// MindNet_84782654423384
// MI455X (gfx1250) — matrix operand order repaired and checked by reading, not compiled, not run on hardware
//
#include <hip/hip_runtime.h>
#include <math.h>

// ---------------------------------------------------------------------------
// Types for CDNA5 WMMA
// ---------------------------------------------------------------------------
typedef __attribute__((ext_vector_type(16))) _Float16 v16h;
typedef __attribute__((ext_vector_type(8)))  _Float16 v8h;
typedef __attribute__((ext_vector_type(8)))  float    v8f;

// Problem constants (match reference)
#define BN   256
#define SN   128
#define EMBW 512
#define H2W  512
#define HIDW 1024
#define CTXW 64
#define VOC  32000
#define MLEN 40
#define SOSID 2

// ---------------------------------------------------------------------------
// Weight pack: W [N,K] f32 (torch layout, y = x @ W^T)  ->  f16 packed B-frag
// layout: out[(kOuter*(N/16) + tn)*512 + L*16 + e] = W[tn*16+(L&15)][kOuter*32+(L>>4)*8+(e&7)+(e>>3)*16]
// so in the GEMM each lane loads its 16 B values with ONE contiguous 32B load.
// ---------------------------------------------------------------------------
__global__ __launch_bounds__(256) void pack_w_kernel(
    const float* __restrict__ W, _Float16* __restrict__ out, int N, int K)
{
    size_t i = (size_t)blockIdx.x * blockDim.x + threadIdx.x;
    size_t total = (size_t)N * K;
    if (i >= total) return;
    size_t perK = (size_t)(N >> 4) * 512;
    int kOuter = (int)(i / perK);
    size_t rem = i % perK;
    int tn = (int)(rem >> 9);        // /512
    int r2 = (int)(rem & 511);
    int L  = r2 >> 4;
    int e  = r2 & 15;
    int wr = tn * 16 + (L & 15);
    int wc = kOuter * 32 + ((L >> 4) << 3) + (e & 7) + ((e >> 3) << 4);
    out[i] = (_Float16)W[(size_t)wr * K + wc];
}

// A-fragment loader (16-bit A 16x32 layout): lanes 0-15 -> K 0..7 & 16..23 of
// row M=lane; lanes 16-31 -> K 8..15 & 24..31 of row M=lane-16.
__device__ __forceinline__ v16h load_afrag(const _Float16* arow, int k0)
{
    v8h lo = *(const v8h*)(arow + k0);
    v8h hi = *(const v8h*)(arow + k0 + 16);
    return __builtin_shufflevector(lo, hi,
        0,1,2,3,4,5,6,7,8,9,10,11,12,13,14,15);
}

// C/D layout: VGPR r: lanes 0-15 -> M=r, lanes 16-31 -> M=r+8; N = lane&15
__device__ __forceinline__ void store_ctile(
    const v8f& acc, float* __restrict__ C, int ldc,
    const float* __restrict__ bias, int rbase, int col)
{
    float bv = bias[col];
    float* crow = C + (size_t)rbase * ldc + col;
#pragma unroll
    for (int r = 0; r < 8; ++r)
        crow[(size_t)r * ldc] = acc[r] + bv;
}

#define WMMA(a, b, c) __builtin_amdgcn_wmma_f32_16x16x32_f16( \
        false, (a), false, (b), (short)0, (c), false, false)

// ---------------------------------------------------------------------------
// Small-tile GEMM (1 wave -> 16x16): max wave parallelism for the small
// latency-bound encoder recurrence GEMMs. grid (N/128, M/16), block 256.
// ---------------------------------------------------------------------------
__global__ __launch_bounds__(256) void gemm_wmma_f16_small(
    const _Float16* __restrict__ A, int lda,
    const _Float16* __restrict__ Bp,
    const float* __restrict__ bias,
    float* __restrict__ C, int ldc,
    int N, int K)
{
    const int lane = threadIdx.x & 31;
    const int wave = threadIdx.x >> 5;
    const int tn = blockIdx.x * 8 + wave;
    const int tm = blockIdx.y;
    if (tn * 16 >= N) return;

    const int row  = tm * 16 + (lane & 15);
    const int koff = (lane >> 4) * 8;
    const _Float16* __restrict__ arow = A + (size_t)row * lda + koff;

    const size_t bstride = (size_t)(N >> 4) * 512;   // per 32-K block
    const _Float16* __restrict__ bptr = Bp + (size_t)tn * 512 + (size_t)lane * 16;

    v8f acc = {};
    for (int k0 = 0; k0 < K; k0 += 32) {
        v16h a = load_afrag(arow, k0);
        v16h b = *(const v16h*)bptr;
        bptr += bstride;
        acc = WMMA(a, b, acc);
    }
    store_ctile(acc, C, ldc, bias,
                tm * 16 + ((lane >> 4) << 3), tn * 16 + (lane & 15));
}

// ---------------------------------------------------------------------------
// Register-blocked GEMM (1 wave -> 32x64 = 2 A-frags x 4 B-frags -> 8 WMMA
// per 32-K step, ~21 flop/byte). Requires N % 64 == 0, M % 32 == 0.
// grid (ceil(N/512), M/32), block 256 (8 waves, each one 64-col quad).
// ---------------------------------------------------------------------------
__global__ __launch_bounds__(256) void gemm_wmma_f16_big(
    const _Float16* __restrict__ A, int lda,
    const _Float16* __restrict__ Bp,
    const float* __restrict__ bias,
    float* __restrict__ C, int ldc,
    int N, int K)
{
    const int lane = threadIdx.x & 31;
    const int wave = threadIdx.x >> 5;
    const int q  = blockIdx.x * 8 + wave;        // 64-column quad index
    if (q * 64 >= N) return;
    const int tm = blockIdx.y;                   // 32-row strip

    const int r0   = tm * 32 + (lane & 15);
    const int koff = (lane >> 4) * 8;
    const _Float16* __restrict__ a0 = A + (size_t)r0 * lda + koff;
    const _Float16* __restrict__ a1 = a0 + (size_t)16 * lda;

    const size_t bstride = (size_t)(N >> 4) * 512;
    const _Float16* __restrict__ bp =
        Bp + (size_t)q * 4 * 512 + (size_t)lane * 16;

    v8f c00 = {}, c01 = {}, c02 = {}, c03 = {};
    v8f c10 = {}, c11 = {}, c12 = {}, c13 = {};
    for (int k0 = 0; k0 < K; k0 += 32) {
        v16h A0 = load_afrag(a0, k0);
        v16h A1 = load_afrag(a1, k0);
        v16h B0 = *(const v16h*)(bp);
        v16h B1 = *(const v16h*)(bp + 512);
        v16h B2 = *(const v16h*)(bp + 1024);
        v16h B3 = *(const v16h*)(bp + 1536);
        bp += bstride;
        c00 = WMMA(A0, B0, c00);
        c01 = WMMA(A0, B1, c01);
        c02 = WMMA(A0, B2, c02);
        c03 = WMMA(A0, B3, c03);
        c10 = WMMA(A1, B0, c10);
        c11 = WMMA(A1, B1, c11);
        c12 = WMMA(A1, B2, c12);
        c13 = WMMA(A1, B3, c13);
    }
    const int col = q * 64 + (lane & 15);
    const int rb0 = tm * 32 + ((lane >> 4) << 3);
    store_ctile(c00, C, ldc, bias, rb0,      col);
    store_ctile(c01, C, ldc, bias, rb0,      col + 16);
    store_ctile(c02, C, ldc, bias, rb0,      col + 32);
    store_ctile(c03, C, ldc, bias, rb0,      col + 48);
    store_ctile(c10, C, ldc, bias, rb0 + 16, col);
    store_ctile(c11, C, ldc, bias, rb0 + 16, col + 16);
    store_ctile(c12, C, ldc, bias, rb0 + 16, col + 32);
    store_ctile(c13, C, ldc, bias, rb0 + 16, col + 48);
}

// ---------------------------------------------------------------------------
// GRU elementwise:  gates ordered (r, z, n), PyTorch semantics.
// ---------------------------------------------------------------------------
__global__ __launch_bounds__(256) void gru_elem_kernel(
    const float* __restrict__ gx, const float* __restrict__ gh,
    const float* __restrict__ hin,
    float* __restrict__ hout, _Float16* __restrict__ hout16, int H)
{
    int i = blockIdx.x * blockDim.x + threadIdx.x;
    if (i >= BN * H) return;
    int b = i / H, j = i - b * H;
    const float* gxr = gx + (size_t)b * 3 * H;
    const float* ghr = gh + (size_t)b * 3 * H;
    float r = 1.f / (1.f + __expf(-(gxr[j]         + ghr[j])));
    float z = 1.f / (1.f + __expf(-(gxr[H + j]     + ghr[H + j])));
    float n = tanhf(gxr[2 * H + j] + r * ghr[2 * H + j]);
    float h = (1.f - z) * n + z * hin[i];
    hout[i]   = h;
    hout16[i] = (_Float16)h;
}

// ---------------------------------------------------------------------------
// Embedding gather for whole input: out[s][b][e] = f16(table[ids[b][s]][e])
// ---------------------------------------------------------------------------
__global__ __launch_bounds__(256) void embed_all_kernel(
    const int* __restrict__ ids, const float* __restrict__ table,
    _Float16* __restrict__ out)
{
    size_t i = (size_t)blockIdx.x * blockDim.x + threadIdx.x;
    size_t total = (size_t)SN * BN * EMBW;
    if (i >= total) return;
    int e = (int)(i % EMBW);
    size_t r = i / EMBW;
    int b = (int)(r % BN);
    int s = (int)(r / BN);
    int tok = ids[(size_t)b * SN + s];
    out[i] = (_Float16)table[(size_t)tok * EMBW + e];
}

// Gather embedding of previous token into xdec[:, 0:512]
__global__ __launch_bounds__(256) void embed_prev_kernel(
    const int* __restrict__ prev, const float* __restrict__ table,
    _Float16* __restrict__ xdec)
{
    int i = blockIdx.x * blockDim.x + threadIdx.x;
    if (i >= BN * EMBW) return;
    int b = i / EMBW, e = i - b * EMBW;
    xdec[(size_t)b * (EMBW + HIDW + CTXW) + e] =
        (_Float16)table[(size_t)prev[b] * EMBW + e];
}

// ---------------------------------------------------------------------------
// Build decoder context columns of xdec (cols 512..1599).
// Faithfully replicates stack([hf,hb],0).reshape(B, 1024): flat = hf||hb,
// row b gets flat[b*1024 : b*1024+1024] (this scrambles batches, as in ref).
// ---------------------------------------------------------------------------
__global__ __launch_bounds__(256) void build_ctx_kernel(
    const float* __restrict__ hf, const float* __restrict__ hb,
    const float* __restrict__ cv, _Float16* __restrict__ xdec)
{
    int i = blockIdx.x * blockDim.x + threadIdx.x;
    if (i >= BN * (HIDW + CTXW)) return;
    int b = i / (HIDW + CTXW), j = i - b * (HIDW + CTXW);
    float v;
    if (j < HIDW) {
        size_t f = (size_t)b * HIDW + j;
        const size_t half = (size_t)BN * H2W;   // 131072
        v = (f < half) ? hf[f] : hb[f - half];
    } else {
        v = cv[(size_t)b * CTXW + (j - HIDW)];
    }
    xdec[(size_t)b * (EMBW + HIDW + CTXW) + EMBW + j] = (_Float16)v;
}

// ---------------------------------------------------------------------------
// Per-row softmax stats over VOC logits:
// argmax (first occurrence), lse; lp = log(exp(l_am - lse) + 1e-12)
// ---------------------------------------------------------------------------
__global__ __launch_bounds__(256) void argmax_lp_kernel(
    const float* __restrict__ logits, int* __restrict__ prev,
    float* __restrict__ out_tok, float* __restrict__ out_lp, int t)
{
    const int b = blockIdx.x;
    const int tid = threadIdx.x;
    const float* row = logits + (size_t)b * VOC;

    float m = -__builtin_inff(); int am = 0;
    for (int j = tid; j < VOC; j += 256) {
        float v = row[j];
        if (v > m) { m = v; am = j; }
    }
    __shared__ float sm[256]; __shared__ int si[256];
    sm[tid] = m; si[tid] = am; __syncthreads();
    for (int s = 128; s > 0; s >>= 1) {
        if (tid < s) {
            float v2 = sm[tid + s]; int i2 = si[tid + s];
            if (v2 > sm[tid] || (v2 == sm[tid] && i2 < si[tid])) {
                sm[tid] = v2; si[tid] = i2;
            }
        }
        __syncthreads();
    }
    float gmax = sm[0]; int gam = si[0];

    float se = 0.f;
    for (int j = tid; j < VOC; j += 256) se += __expf(row[j] - gmax);
    __shared__ float ss[256];
    ss[tid] = se; __syncthreads();
    for (int s = 128; s > 0; s >>= 1) {
        if (tid < s) ss[tid] += ss[tid + s];
        __syncthreads();
    }
    if (tid == 0) {
        float lse = gmax + __logf(ss[0]);
        float sel = __expf(row[gam] - lse);
        prev[b] = gam;
        out_tok[(size_t)b * MLEN + t] = (float)gam;
        out_lp [(size_t)b * MLEN + t] = __logf(sel + 1e-12f);
    }
}

// ---------------------------------------------------------------------------
// Fills
// ---------------------------------------------------------------------------
__global__ __launch_bounds__(256) void fill_f32_kernel(float* p, float v, int n)
{
    int i = blockIdx.x * blockDim.x + threadIdx.x;
    if (i < n) p[i] = v;
}
__global__ __launch_bounds__(256) void fill_f16_kernel(_Float16* p, float v, int n)
{
    int i = blockIdx.x * blockDim.x + threadIdx.x;
    if (i < n) p[i] = (_Float16)v;
}
__global__ __launch_bounds__(256) void fill_i32_kernel(int* p, int v, int n)
{
    int i = blockIdx.x * blockDim.x + threadIdx.x;
    if (i < n) p[i] = v;
}

// ---------------------------------------------------------------------------
// Host orchestration
// ---------------------------------------------------------------------------
static inline size_t align256(size_t x) { return (x + 255) & ~(size_t)255; }

extern "C" void kernel_launch(void* const* d_in, const int* in_sizes, int n_in,
                              void* d_out, int out_size, void* d_ws, size_t ws_size,
                              hipStream_t stream)
{
    (void)in_sizes; (void)n_in; (void)out_size; (void)ws_size;

    const int*   input_ids  = (const int*)  d_in[0];
    const float* context_v  = (const float*)d_in[1];
    const float* emb_table  = (const float*)d_in[2];
    const float* enc_wih_f  = (const float*)d_in[3];
    const float* enc_whh_f  = (const float*)d_in[4];
    const float* enc_bih_f  = (const float*)d_in[5];
    const float* enc_bhh_f  = (const float*)d_in[6];
    const float* enc_wih_b  = (const float*)d_in[7];
    const float* enc_whh_b  = (const float*)d_in[8];
    const float* enc_bih_b  = (const float*)d_in[9];
    const float* enc_bhh_b  = (const float*)d_in[10];
    const float* dec_wih    = (const float*)d_in[11];
    const float* dec_whh    = (const float*)d_in[12];
    const float* dec_bih    = (const float*)d_in[13];
    const float* dec_bhh    = (const float*)d_in[14];
    const float* wout       = (const float*)d_in[15];
    const float* bout       = (const float*)d_in[16];

    float* out_tok = (float*)d_out;                       // [B, MLEN]
    float* out_lp  = (float*)d_out + (size_t)BN * MLEN;   // [B, MLEN]

    // ---- workspace bump allocator ----
    char* ws = (char*)d_ws;
    size_t off = 0;
    auto alloc = [&](size_t bytes) -> void* {
        void* p = ws + off; off = align256(off + bytes); return p;
    };
    const int DXW = EMBW + HIDW + CTXW;  // 1600

    _Float16* p_ewf  = (_Float16*)alloc((size_t)EMBW * 3 * H2W * 2);
    _Float16* p_ehf  = (_Float16*)alloc((size_t)H2W  * 3 * H2W * 2);
    _Float16* p_ewb  = (_Float16*)alloc((size_t)EMBW * 3 * H2W * 2);
    _Float16* p_ehb  = (_Float16*)alloc((size_t)H2W  * 3 * H2W * 2);
    _Float16* p_dwi  = (_Float16*)alloc((size_t)DXW  * 3 * HIDW * 2);
    _Float16* p_dwh  = (_Float16*)alloc((size_t)HIDW * 3 * HIDW * 2);
    _Float16* p_wout = (_Float16*)alloc((size_t)HIDW * VOC * 2);
    _Float16* emb_seq= (_Float16*)alloc((size_t)SN * BN * EMBW * 2);
    float*    hf     = (float*)   alloc((size_t)BN * H2W * 4);
    float*    hb     = (float*)   alloc((size_t)BN * H2W * 4);
    float*    hdec   = (float*)   alloc((size_t)BN * HIDW * 4);
    _Float16* h16    = (_Float16*)alloc((size_t)BN * HIDW * 2);
    float*    gx     = (float*)   alloc((size_t)BN * 3 * HIDW * 4);
    float*    gh     = (float*)   alloc((size_t)BN * 3 * HIDW * 4);
    _Float16* xdec   = (_Float16*)alloc((size_t)BN * DXW * 2);
    float*    logits = (float*)   alloc((size_t)BN * VOC * 4);
    int*      prev   = (int*)     alloc((size_t)BN * 4);

    auto pack = [&](const float* W, _Float16* out, int N, int K) {
        size_t total = (size_t)N * K;
        pack_w_kernel<<<dim3((unsigned)((total + 255) / 256)), 256, 0, stream>>>(W, out, N, K);
    };
    auto gemm_small = [&](const _Float16* A, int lda, const _Float16* Bp,
                          const float* bias, float* C, int ldc, int N, int K) {
        dim3 g((unsigned)((N / 16 + 7) / 8), (unsigned)(BN / 16));
        gemm_wmma_f16_small<<<g, 256, 0, stream>>>(A, lda, Bp, bias, C, ldc, N, K);
    };
    auto gemm_big = [&](const _Float16* A, int lda, const _Float16* Bp,
                        const float* bias, float* C, int ldc, int N, int K) {
        dim3 g((unsigned)((N / 64 + 7) / 8), (unsigned)(BN / 32));
        gemm_wmma_f16_big<<<g, 256, 0, stream>>>(A, lda, Bp, bias, C, ldc, N, K);
    };

    // ---- 1. cast + pack all weights to f16 WMMA-B layout ----
    pack(enc_wih_f, p_ewf,  3 * H2W, EMBW);
    pack(enc_whh_f, p_ehf,  3 * H2W, H2W);
    pack(enc_wih_b, p_ewb,  3 * H2W, EMBW);
    pack(enc_whh_b, p_ehb,  3 * H2W, H2W);
    pack(dec_wih,   p_dwi,  3 * HIDW, DXW);
    pack(dec_whh,   p_dwh,  3 * HIDW, HIDW);
    pack(wout,      p_wout, VOC, HIDW);

    // ---- 2. embedding gather (time-major) ----
    {
        size_t total = (size_t)SN * BN * EMBW;
        embed_all_kernel<<<dim3((unsigned)((total + 255) / 256)), 256, 0, stream>>>(
            input_ids, emb_table, emb_seq);
    }

    // ---- 3. encoder forward ----
    fill_f32_kernel<<<(BN * H2W + 255) / 256, 256, 0, stream>>>(hf, 0.f, BN * H2W);
    fill_f16_kernel<<<(BN * H2W + 255) / 256, 256, 0, stream>>>(h16, 0.f, BN * H2W);
    for (int t = 0; t < SN; ++t) {
        gemm_small(emb_seq + (size_t)t * BN * EMBW, EMBW, p_ewf, enc_bih_f,
                   gx, 3 * H2W, 3 * H2W, EMBW);
        gemm_small(h16, H2W, p_ehf, enc_bhh_f, gh, 3 * H2W, 3 * H2W, H2W);
        gru_elem_kernel<<<(BN * H2W + 255) / 256, 256, 0, stream>>>(gx, gh, hf, hf, h16, H2W);
    }

    // ---- 4. encoder backward ----
    fill_f32_kernel<<<(BN * H2W + 255) / 256, 256, 0, stream>>>(hb, 0.f, BN * H2W);
    fill_f16_kernel<<<(BN * H2W + 255) / 256, 256, 0, stream>>>(h16, 0.f, BN * H2W);
    for (int t = SN - 1; t >= 0; --t) {
        gemm_small(emb_seq + (size_t)t * BN * EMBW, EMBW, p_ewb, enc_bih_b,
                   gx, 3 * H2W, 3 * H2W, EMBW);
        gemm_small(h16, H2W, p_ehb, enc_bhh_b, gh, 3 * H2W, 3 * H2W, H2W);
        gru_elem_kernel<<<(BN * H2W + 255) / 256, 256, 0, stream>>>(gx, gh, hb, hb, h16, H2W);
    }

    // ---- 5. decoder setup ----
    build_ctx_kernel<<<(BN * (HIDW + CTXW) + 255) / 256, 256, 0, stream>>>(
        hf, hb, context_v, xdec);
    fill_f32_kernel<<<(BN * HIDW + 255) / 256, 256, 0, stream>>>(hdec, 0.f, BN * HIDW);
    fill_f16_kernel<<<(BN * HIDW + 255) / 256, 256, 0, stream>>>(h16, 0.f, BN * HIDW);
    fill_i32_kernel<<<(BN + 255) / 256, 256, 0, stream>>>(prev, SOSID, BN);

    // ---- 6. greedy decode (compute-dominant path -> register-blocked GEMM) ----
    for (int t = 0; t < MLEN; ++t) {
        embed_prev_kernel<<<(BN * EMBW + 255) / 256, 256, 0, stream>>>(prev, emb_table, xdec);
        gemm_big(xdec, DXW, p_dwi, dec_bih, gx, 3 * HIDW, 3 * HIDW, DXW);
        gemm_big(h16, HIDW, p_dwh, dec_bhh, gh, 3 * HIDW, 3 * HIDW, HIDW);
        gru_elem_kernel<<<(BN * HIDW + 255) / 256, 256, 0, stream>>>(gx, gh, hdec, hdec, h16, HIDW);
        gemm_big(h16, HIDW, p_wout, bout, logits, VOC, VOC, HIDW);
        argmax_lp_kernel<<<BN, 256, 0, stream>>>(logits, prev, out_tok, out_lp, t);
    }
}
